// referenceAttImg_88132728914467
// MI455X (gfx1250) — matrix operand order repaired and checked by reading, not compiled, not run on hardware
//
#include <hip/hip_runtime.h>
#include <hip/hip_bf16.h>

#define BATCH 8
#define SEQ   1024
#define DIM   1024

typedef __attribute__((ext_vector_type(16))) _Float16 v16h;
typedef __attribute__((ext_vector_type(8)))  _Float16 v8h;
typedef __attribute__((ext_vector_type(4)))  _Float16 v4h;
typedef __attribute__((ext_vector_type(8)))  float    v8f;

__device__ __forceinline__ int lane_id() { return threadIdx.x & 31; }

// ---------------------------------------------------------------------------
// WMMA fragment loaders for v_wmma_f32_16x16x32_f16, wave32 (this file's contiguous k order, k = 16h + e)
//
// A (16x32 f16, M x K): lane l -> row m = l&15, half h = l>>4.
//   vector elems 0..7  = A[m, k0 + 16h + 0..7]     (16B contiguous)
//   vector elems 8..15 = A[m, k0 + 16h + 8..15]    (16B contiguous)
//
// B (32x16 f16, K x N) where B[k][n] = Rows[n][k] (multiply-by-transpose):
//   lane l -> col n = l&15, half h = l>>4; elems e = Rows[n0+n][k0 + 16h + e]
//   -> one contiguous 32B load per lane.
//
// C/D (16x16 f32): VGPR r, lane l -> D[r + 8*(l>>4)][l&15]
// ---------------------------------------------------------------------------

__device__ __forceinline__ v16h load_a_f16(const _Float16* A, int ld, int row0, int k0) {
    int l = lane_id();
    int m = l & 15, h = l >> 4;
    const _Float16* p = A + (size_t)(row0 + m) * ld + k0 + h * 16;
    v8h lo = *(const v8h*)p;
    v8h hi = *(const v8h*)(p + 8);
    v16h r;
#pragma unroll
    for (int i = 0; i < 8; ++i) { r[i] = lo[i]; r[8 + i] = hi[i]; }
    return r;
}

__device__ __forceinline__ v16h load_a_f32(const float* A, int ld, int row0, int k0) {
    int l = lane_id();
    int m = l & 15, h = l >> 4;
    const float* p = A + (size_t)(row0 + m) * ld + k0 + h * 16;
    v16h r;
#pragma unroll
    for (int i = 0; i < 8; ++i) { r[i] = (_Float16)p[i]; r[8 + i] = (_Float16)p[8 + i]; }
    return r;
}

__device__ __forceinline__ v16h load_bT_f16(const _Float16* Rows, int ld, int n0, int k0) {
    int l = lane_id();
    int n = l & 15, h = l >> 4;
    const _Float16* p = Rows + (size_t)(n0 + n) * ld + k0 + h * 16;
    return *(const v16h*)p;          // 32B aligned: k0 % 32 == 0, h*16 halves = 32B
}

// ---------------------------------------------------------------------------
// Kernel 0: one-shot f32 -> f16 conversion of the three weight matrices.
// Memory bound: ~18 MB traffic, < 1 us. Removes 512x-redundant cvts from the
// projection hot loop.
// ---------------------------------------------------------------------------
__global__ __launch_bounds__(256) void cvt_w_kernel(
    const float* __restrict__ Wq, const float* __restrict__ Wk,
    const float* __restrict__ Wv,
    _Float16* __restrict__ Wqh, _Float16* __restrict__ Wkh,
    _Float16* __restrict__ Wvh)
{
    int which = blockIdx.y;
    const float* s = (which == 0) ? Wq : (which == 1) ? Wk : Wv;
    _Float16*    d = (which == 0) ? Wqh : (which == 1) ? Wkh : Wvh;
    int i = (blockIdx.x * 256 + threadIdx.x) * 4;           // DIM*DIM = 1M elems
    float4 v = *(const float4*)(s + i);
    v4h o;
    o[0] = (_Float16)v.x; o[1] = (_Float16)v.y;
    o[2] = (_Float16)v.z; o[3] = (_Float16)v.w;
    *(v4h*)(d + i) = o;
}

// ---------------------------------------------------------------------------
// Kernel 1: QKV projections  out[l,n] = sum_k X[l,k]*W[n,k] + b[n]
// blockIdx.y: 0 -> Q (rc -> Qh), 1 -> K (ri -> Kh), 2 -> V (ri -> Vt, [D,SEQ])
// One wave computes a 32x64 output block (2x4 tiles, 8 WMMAs per k-step):
// A-frags reused 4x, B-frags 2x; B operand is pre-converted f16.
// ---------------------------------------------------------------------------
__global__ __launch_bounds__(256) void proj_kernel(
    const float* __restrict__ rc, const float* __restrict__ ri,
    const _Float16* __restrict__ Wqh, const _Float16* __restrict__ Wkh,
    const _Float16* __restrict__ Wvh,
    const float* __restrict__ bq, const float* __restrict__ bk,
    const float* __restrict__ bv,
    _Float16* __restrict__ Qh, _Float16* __restrict__ Kh, _Float16* __restrict__ Vt)
{
    int mode = blockIdx.y;
    const float* X; const _Float16* W; const float* bias; _Float16* out;
    if (mode == 0)      { X = rc; W = Wqh; bias = bq; out = Qh; }
    else if (mode == 1) { X = ri; W = Wkh; bias = bk; out = Kh; }
    else                { X = ri; W = Wvh; bias = bv; out = Vt; }

    int wave = threadIdx.x >> 5;
    int gw   = blockIdx.x * 8 + wave;          // 0 .. 4095
    int b    = gw >> 9;                        // 512 waves per batch
    int t    = gw & 511;                       // 32 row-blocks x 16 col-blocks
    int row0 = (t >> 4) << 5;                  // 32-row block
    int col0 = (t & 15) << 6;                  // 64-col block

    const float* Xb = X + (size_t)b * SEQ * DIM;

    v8f acc[2][4] = {};
    for (int k0 = 0; k0 < DIM; k0 += 32) {
        v16h a[2];
#pragma unroll
        for (int i = 0; i < 2; ++i) a[i] = load_a_f32(Xb, DIM, row0 + 16 * i, k0);
        v16h bt[4];
#pragma unroll
        for (int j = 0; j < 4; ++j) bt[j] = load_bT_f16(W, DIM, col0 + 16 * j, k0);
#pragma unroll
        for (int i = 0; i < 2; ++i)
#pragma unroll
            for (int j = 0; j < 4; ++j)
                acc[i][j] = __builtin_amdgcn_wmma_f32_16x16x32_f16(
                    false, a[i], false, bt[j], (short)0, acc[i][j], false, false);
    }

    int l = lane_id();
    int n = l & 15, h = l >> 4;

    if (mode != 2) {
        _Float16* O = out + (size_t)b * SEQ * DIM;
#pragma unroll
        for (int i = 0; i < 2; ++i) {
#pragma unroll
            for (int j = 0; j < 4; ++j) {
                float bval = bias[col0 + 16 * j + n];
#pragma unroll
                for (int r = 0; r < 8; ++r) {
                    int m = row0 + 16 * i + r + 8 * h;
                    O[(size_t)m * DIM + col0 + 16 * j + n] = (_Float16)(acc[i][j][r] + bval);
                }
            }
        }
    } else {
        // transposed store: Vt[b][d][l], 16B packed per (i,j)
#pragma unroll
        for (int i = 0; i < 2; ++i) {
#pragma unroll
            for (int j = 0; j < 4; ++j) {
                float bval = bias[col0 + 16 * j + n];
                _Float16* O = out + (size_t)b * DIM * SEQ
                                  + (size_t)(col0 + 16 * j + n) * SEQ
                                  + row0 + 16 * i + 8 * h;
                v8h pk;
#pragma unroll
                for (int r = 0; r < 8; ++r) pk[r] = (_Float16)(acc[i][j][r] + bval);
                *(v8h*)O = pk;
            }
        }
    }
}

// ---------------------------------------------------------------------------
// Kernel 2: logits = Q K^T, scaled by 1/sqrt(K[b,q,k]), row softmax -> attn.
// One block per (batch, 16-row q tile). Wave w owns columns [128w, 128w+128):
// shared A-frag, 8 B tiles, pure-f16 hot loop. Logits staged in 64KB LDS.
// ---------------------------------------------------------------------------
__global__ __launch_bounds__(256) void qk_softmax_kernel(
    const _Float16* __restrict__ Qh, const _Float16* __restrict__ Kh,
    float* __restrict__ attn)
{
    __shared__ float slog[16 * SEQ];           // 64 KB

    int wg   = blockIdx.x;                     // 0 .. 511
    int b    = wg >> 6;
    int q0   = (wg & 63) << 4;
    int wave = threadIdx.x >> 5;
    int l    = lane_id();
    int n    = l & 15, h = l >> 4;

    const _Float16* Qb = Qh + (size_t)b * SEQ * DIM;
    const _Float16* Kb = Kh + (size_t)b * SEQ * DIM;

    v8f acc[8] = {};
    int n0base = wave * 128;
    for (int k0 = 0; k0 < DIM; k0 += 32) {
        v16h a = load_a_f16(Qb, DIM, q0, k0);
#pragma unroll
        for (int t = 0; t < 8; ++t) {
            v16h bt = load_bT_f16(Kb, DIM, n0base + t * 16, k0);
            acc[t] = __builtin_amdgcn_wmma_f32_16x16x32_f16(false, a, false, bt,
                                                            (short)0, acc[t], false, false);
        }
    }

#pragma unroll
    for (int t = 0; t < 8; ++t) {
        int col = n0base + t * 16 + n;
#pragma unroll
        for (int r = 0; r < 8; ++r)
            slog[(r + 8 * h) * SEQ + col] = acc[t][r];
    }
    __syncthreads();

    // softmax over k for rows q = 2*wave, 2*wave+1
#pragma unroll
    for (int rr = 0; rr < 2; ++rr) {
        int q = wave * 2 + rr;
        const _Float16* Krow = Kb + (size_t)(q0 + q) * DIM;
        float* row = slog + q * SEQ;

        float mx = -3.0e38f;
        for (int c = l; c < SEQ; c += 32) {
            float kv = (float)Krow[c];
            float v  = row[c] * __frsqrt_rn(kv);
            row[c] = v;
            mx = fmaxf(mx, v);
        }
#pragma unroll
        for (int off = 16; off >= 1; off >>= 1) mx = fmaxf(mx, __shfl_xor(mx, off, 32));

        float sum = 0.0f;
        for (int c = l; c < SEQ; c += 32) {
            float e = __expf(row[c] - mx);
            row[c] = e;
            sum += e;
        }
#pragma unroll
        for (int off = 16; off >= 1; off >>= 1) sum += __shfl_xor(sum, off, 32);

        float inv = 1.0f / sum;
        float* orow = attn + (size_t)b * SEQ * SEQ + (size_t)(q0 + q) * SEQ;
        for (int c = l; c < SEQ; c += 32) orow[c] = row[c] * inv;
    }
}

// ---------------------------------------------------------------------------
// Kernel 3: out[b,q,d] = sum_k attn[b,q,k] * V[b,k,d]   (V stored as Vt[b,d,k])
// One wave computes a 32x64 output block (2x4 tiles), same blocking as proj.
// ---------------------------------------------------------------------------
__global__ __launch_bounds__(256) void attn_v_kernel(
    const float* __restrict__ attn, const _Float16* __restrict__ Vt,
    float* __restrict__ out)
{
    int wave = threadIdx.x >> 5;
    int gw   = blockIdx.x * 8 + wave;          // 0 .. 4095
    int b    = gw >> 9;
    int t    = gw & 511;
    int q0   = (t >> 4) << 5;
    int d0   = (t & 15) << 6;

    const float*    Ab = attn + (size_t)b * SEQ * SEQ;
    const _Float16* Vb = Vt   + (size_t)b * DIM * SEQ;

    v8f acc[2][4] = {};
    for (int k0 = 0; k0 < SEQ; k0 += 32) {
        v16h a[2];
#pragma unroll
        for (int i = 0; i < 2; ++i) a[i] = load_a_f32(Ab, SEQ, q0 + 16 * i, k0);
        v16h bt[4];
#pragma unroll
        for (int j = 0; j < 4; ++j) bt[j] = load_bT_f16(Vb, SEQ, d0 + 16 * j, k0);
#pragma unroll
        for (int i = 0; i < 2; ++i)
#pragma unroll
            for (int j = 0; j < 4; ++j)
                acc[i][j] = __builtin_amdgcn_wmma_f32_16x16x32_f16(
                    false, a[i], false, bt[j], (short)0, acc[i][j], false, false);
    }

    int l = lane_id();
    int n = l & 15, h = l >> 4;
    float* O = out + (size_t)b * SEQ * DIM;
#pragma unroll
    for (int i = 0; i < 2; ++i)
#pragma unroll
        for (int j = 0; j < 4; ++j)
#pragma unroll
            for (int r = 0; r < 8; ++r)
                O[(size_t)(q0 + 16 * i + r + 8 * h) * DIM + d0 + 16 * j + n] = acc[i][j][r];
}

// ---------------------------------------------------------------------------
extern "C" void kernel_launch(void* const* d_in, const int* in_sizes, int n_in,
                              void* d_out, int out_size, void* d_ws, size_t ws_size,
                              hipStream_t stream)
{
    const float* rc = (const float*)d_in[0];   // reference_caption [8,1024,1024]
    const float* ri = (const float*)d_in[1];   // reference_image   [8,1024,1024]
    // d_in[2] target_caption: unused by the reference computation
    const float* Wq = (const float*)d_in[3];
    const float* bq = (const float*)d_in[4];
    const float* Wk = (const float*)d_in[5];
    const float* bk = (const float*)d_in[6];
    const float* Wv = (const float*)d_in[7];
    const float* bv = (const float*)d_in[8];

    float* out  = (float*)d_out;                              // [8,1024,1024]
    float* attn = out + (size_t)BATCH * SEQ * SEQ;            // [8,1024,1024]

    _Float16* Qh  = (_Float16*)d_ws;                          // 16 MB
    _Float16* Kh  = Qh  + (size_t)BATCH * SEQ * DIM;          // 16 MB
    _Float16* Vt  = Kh  + (size_t)BATCH * SEQ * DIM;          // 16 MB (transposed)
    _Float16* Wqh = Vt  + (size_t)BATCH * SEQ * DIM;          // 2 MB
    _Float16* Wkh = Wqh + (size_t)DIM * DIM;                  // 2 MB
    _Float16* Wvh = Wkh + (size_t)DIM * DIM;                  // 2 MB

    dim3 blk(256);
    cvt_w_kernel<<<dim3(DIM * DIM / 1024, 3), blk, 0, stream>>>(Wq, Wk, Wv,
                                                                Wqh, Wkh, Wvh);
    proj_kernel<<<dim3(512, 3), blk, 0, stream>>>(rc, ri, Wqh, Wkh, Wvh,
                                                  bq, bk, bv, Qh, Kh, Vt);
    qk_softmax_kernel<<<dim3(512), blk, 0, stream>>>(Qh, Kh, attn);
    attn_v_kernel<<<dim3(512), blk, 0, stream>>>(attn, Vt, out);
}
